// xLSTM_31756988187114
// MI455X (gfx1250) — matrix operand order repaired and checked by reading, not compiled, not run on hardware
//
#include <hip/hip_runtime.h>
#include <hip/hip_bf16.h>

// ---------------- model dims ----------------
#define Bc   8
#define Sc   512
#define Ec   512
#define Ic   1024
#define NHc  4
#define KWc  4
#define DHSc 128
#define DHMc 256
#define Pc   704
#define Lc   2
#define EPSc    1e-6f
#define LNEPSc  1e-5f

typedef __bf16 bf16_t;
typedef __attribute__((ext_vector_type(2)))  __bf16 bf16x2;
typedef __attribute__((ext_vector_type(2)))  float  f32x2;
typedef __attribute__((ext_vector_type(16))) __bf16 v16bf;
typedef __attribute__((ext_vector_type(8)))  float  v8f;

union FragV { v16bf v; uint4 q[2]; };
union Pack4 { bf16x2 h[2]; uint2 u; };

// native f32 -> bf16 conversion (RNE); lowers to v_cvt_pk_bf16_f32
__device__ __forceinline__ bf16_t f2bf(float f) { return (bf16_t)f; }

__device__ __forceinline__ uint2 pack4bf(float4 v) {
  Pack4 p;
  p.h[0] = __builtin_convertvector((f32x2){v.x, v.y}, bf16x2);  // one v_cvt_pk_bf16_f32
  p.h[1] = __builtin_convertvector((f32x2){v.z, v.w}, bf16x2);  // one v_cvt_pk_bf16_f32
  return p.u;
}
__device__ __forceinline__ float sigmf(float x)  { return 1.f / (1.f + expf(-x)); }
__device__ __forceinline__ float swishf(float x) { return x / (1.f + expf(-x)); }
__device__ __forceinline__ float logsigf(float x){ return fminf(x, 0.f) - log1pf(expf(-fabsf(x))); }
__device__ __forceinline__ float geluf(float x)  { return 0.5f * x * (1.f + erff(x * 0.70710678118654752f)); }

// ---------------------------------------------------------------------------
// Tiled bf16 WMMA GEMM:  C[M,N] (f32) = A[M,K] (f32->bf16) * B (f32->bf16)
// REQUIRES: M % 128 == 0, N % 128 == 0, K % 32 == 0, lda/ldb % 4 == 0,
//           batch offsets % 4 == 0 (true for every call in this model).
// B is (K,N) row-major, or (N,K) row-major if flags&1 (wt).
// flags: 1=B transposed(N,K)  2=add bias[col]  4=accumulate into C (residual)
// Batch z: off = (z/dv)*s1 + (z%dv)*s2 for A,B,C independently.
// Tile: BM=128 BN=128 BK=32; 256 threads = 8 waves (4x2); wave = 32x64 = 8 wmma.
// Register-prefetch of the next K-tile overlaps global loads with WMMA.
// ---------------------------------------------------------------------------
__global__ void __launch_bounds__(256) gemm_bf16_wmma(
    const float* __restrict__ A, const float* __restrict__ Bw,
    const float* __restrict__ bias, float* __restrict__ C,
    int M, int N, int Kd, int lda, int ldb, int ldc,
    int dv, long long sA1, long long sA2, long long sB1, long long sB2,
    long long sC1, long long sC2, int flags)
{
  const int wt   = flags & 1;
  const int hasB = flags & 2;
  const int accF = flags & 4;
  const int z = blockIdx.z;
  const long long offA = (long long)(z / dv) * sA1 + (long long)(z % dv) * sA2;
  const long long offB = (long long)(z / dv) * sB1 + (long long)(z % dv) * sB2;
  const long long offC = (long long)(z / dv) * sC1 + (long long)(z % dv) * sC2;
  const int rowBase = blockIdx.y * 128;
  const int colBase = blockIdx.x * 128;

  __shared__ __align__(16) bf16_t As[128 * 40]; // [m][k], rows padded 32->40
  __shared__ __align__(16) bf16_t Bs[128 * 40]; // [n][k], rows padded 32->40

  const int tid  = threadIdx.x;
  const int wave = tid >> 5;
  const int lane = tid & 31;
  const int wm = wave & 3, wn = wave >> 2;    // 4 x 2 wave grid
  const int h  = lane >> 4, lr = lane & 15;

  v8f zero = {0.f,0.f,0.f,0.f,0.f,0.f,0.f,0.f};
  v8f acc[2][4];
  #pragma unroll
  for (int i = 0; i < 2; i++)
    #pragma unroll
    for (int j = 0; j < 4; j++) acc[i][j] = zero;

  float4 ar[4], br[4];

  // ---- tile load helpers (exact-fit, unguarded, vectorized) ----
  auto loadA = [&](int k0) {
    #pragma unroll
    for (int i = 0; i < 4; i++) {
      int t = tid + i * 256;                  // 1024 float4 = 128x32 floats
      int m = t >> 3, q = t & 7;
      ar[i] = *(const float4*)&A[offA + (long long)(rowBase + m) * lda + k0 + 4 * q];
    }
  };
  auto loadB = [&](int k0) {
    if (wt) {
      #pragma unroll
      for (int i = 0; i < 4; i++) {
        int t = tid + i * 256;
        int n = t >> 3, q = t & 7;
        br[i] = *(const float4*)&Bw[offB + (long long)(colBase + n) * ldb + k0 + 4 * q];
      }
    } else {
      #pragma unroll
      for (int i = 0; i < 4; i++) {
        int t = tid + i * 256;                // 32 k-rows x 32 float4 along n
        int kk = t >> 5, q = t & 31;
        br[i] = *(const float4*)&Bw[offB + (long long)(k0 + kk) * ldb + colBase + 4 * q];
      }
    }
  };
  auto storeTiles = [&]() {
    #pragma unroll
    for (int i = 0; i < 4; i++) {
      int t = tid + i * 256;
      int m = t >> 3, q = t & 7;
      *(uint2*)&As[m * 40 + 4 * q] = pack4bf(ar[i]);
    }
    if (wt) {
      #pragma unroll
      for (int i = 0; i < 4; i++) {
        int t = tid + i * 256;
        int n = t >> 3, q = t & 7;
        *(uint2*)&Bs[n * 40 + 4 * q] = pack4bf(br[i]);
      }
    } else {
      #pragma unroll
      for (int i = 0; i < 4; i++) {
        int t = tid + i * 256;
        int kk = t >> 5, q = t & 31;
        int n0 = 4 * q;
        Bs[(n0 + 0) * 40 + kk] = f2bf(br[i].x);
        Bs[(n0 + 1) * 40 + kk] = f2bf(br[i].y);
        Bs[(n0 + 2) * 40 + kk] = f2bf(br[i].z);
        Bs[(n0 + 3) * 40 + kk] = f2bf(br[i].w);
      }
    }
  };

  loadA(0);
  loadB(0);
  for (int k0 = 0; k0 < Kd; k0 += 32) {
    storeTiles();
    __syncthreads();
    if (k0 + 32 < Kd) { loadA(k0 + 32); loadB(k0 + 32); }  // prefetch next tile

    FragV fa[2], fb[4];
    #pragma unroll
    for (int i = 0; i < 2; i++) {
      int arr = wm * 32 + i * 16 + lr;
      fa[i].q[0] = *(const uint4*)&As[arr * 40 + 8 * h];      // K = 8h..8h+7
      fa[i].q[1] = *(const uint4*)&As[arr * 40 + 16 + 8 * h]; // K = 16+8h..
    }
    #pragma unroll
    for (int j = 0; j < 4; j++) {
      int bc = wn * 64 + j * 16 + lr;
      fb[j].q[0] = *(const uint4*)&Bs[bc * 40 + 8 * h];       // K = 8h..8h+7
      fb[j].q[1] = *(const uint4*)&Bs[bc * 40 + 16 + 8 * h];  // K = 16+8h..
    }
    #pragma unroll
    for (int i = 0; i < 2; i++)
      #pragma unroll
      for (int j = 0; j < 4; j++)
        acc[i][j] = __builtin_amdgcn_wmma_f32_16x16x32_bf16(
            false, fa[i].v, false, fb[j].v, (short)0, acc[i][j], false, false);
    __syncthreads();
  }

  #pragma unroll
  for (int i = 0; i < 2; i++)
    #pragma unroll
    for (int j = 0; j < 4; j++) {
      int rT = rowBase + wm * 32 + i * 16;
      int cT = colBase + wn * 64 + j * 16;
      int c  = cT + lr;
      float bv = hasB ? bias[c] : 0.f;
      #pragma unroll
      for (int e = 0; e < 8; e++) {
        int r = rT + h * 8 + e;
        float v = acc[i][j][e] + bv;
        long long idx = offC + (long long)r * ldc + c;
        if (accF) v += C[idx];
        C[idx] = v;
      }
    }
}

// ------------------------- LayerNorm (weight only) -------------------------
__global__ void ln_kernel(const float* __restrict__ x, const float* __restrict__ w,
                          float* __restrict__ y, int N)
{
  long long row = blockIdx.x;
  const float* xr = x + row * N;
  float* yr = y + row * N;
  __shared__ float r1[256], r2[256];
  float s = 0.f, s2 = 0.f;
  for (int c = threadIdx.x; c < N; c += 256) { float v = xr[c]; s += v; s2 += v * v; }
  r1[threadIdx.x] = s; r2[threadIdx.x] = s2;
  __syncthreads();
  for (int st = 128; st > 0; st >>= 1) {
    if (threadIdx.x < st) { r1[threadIdx.x] += r1[threadIdx.x + st]; r2[threadIdx.x] += r2[threadIdx.x + st]; }
    __syncthreads();
  }
  float mu = r1[0] / N;
  float var = r2[0] / N - mu * mu;
  float rs = rsqrtf(var + LNEPSc);
  for (int c = threadIdx.x; c < N; c += 256)
    yr[c] = (xr[c] - mu) * rs * w[c];
}

// ------------------- depthwise causal conv (K=4) + swish -------------------
__global__ void conv_swish_kernel(const float* __restrict__ x, const float* __restrict__ w,
                                  const float* __restrict__ b, float* __restrict__ y,
                                  int C, int ldx, long long total)
{
  long long idx = (long long)blockIdx.x * blockDim.x + threadIdx.x;
  if (idx >= total) return;
  int c = (int)(idx % C);
  long long t = idx / C;
  int s = (int)(t % Sc);
  long long bb = t / Sc;
  float acc = b[c];
  #pragma unroll
  for (int j = 0; j < KWc; j++) {
    int ss = s + j - (KWc - 1);
    if (ss >= 0) acc += x[(bb * Sc + ss) * ldx + c] * w[c * KWc + j];
  }
  y[idx] = swishf(acc);
}

// -------------- block-diagonal 4x4 q/k/v projections (one pass) -------------
__global__ void qkv_kernel(const float* __restrict__ xc, int ldxc,
                           const float* __restrict__ xm, int ldxm,
                           const float* __restrict__ qw, const float* __restrict__ kw,
                           const float* __restrict__ vw,
                           float* __restrict__ q, float* __restrict__ k, float* __restrict__ v)
{
  long long idx = (long long)blockIdx.x * blockDim.x + threadIdx.x;
  if (idx >= (long long)Bc * Sc * Ic) return;
  int c = (int)(idx & (Ic - 1));
  long long row = idx >> 10;
  int nb = c >> 2, j = c & 3;
  const float* xr  = xc + row * ldxc + (nb << 2);
  const float* xmr = xm + row * ldxm + (nb << 2);
  float aq = 0.f, ak = 0.f, av = 0.f;
  #pragma unroll
  for (int i = 0; i < 4; i++) {
    int wi = (nb * 4 + j) * 4 + i;
    aq += xr[i]  * qw[wi];
    ak += xr[i]  * kw[wi];
    av += xmr[i] * vw[wi];
  }
  q[idx] = aq; k[idx] = ak; v[idx] = av;
}

// ------------- ig/fg projections: (B*S,3I)@(3I,4)+b -> (B,NH,S) -------------
__global__ void igfg_kernel(const float* __restrict__ q, const float* __restrict__ k,
                            const float* __restrict__ v,
                            const float* __restrict__ igw, const float* __restrict__ igb,
                            const float* __restrict__ fgw, const float* __restrict__ fgb,
                            float* __restrict__ ig, float* __restrict__ fg)
{
  int row = blockIdx.x;            // b*S + s
  int b = row / Sc, s = row % Sc;
  int tid = threadIdx.x;           // 128
  float ai[4] = {0,0,0,0}, af[4] = {0,0,0,0};
  for (int kk = tid; kk < 3 * Ic; kk += 128) {
    float x = (kk < Ic) ? q[(long long)row * Ic + kk]
            : (kk < 2 * Ic) ? k[(long long)row * Ic + kk - Ic]
            : v[(long long)row * Ic + kk - 2 * Ic];
    #pragma unroll
    for (int hh = 0; hh < 4; hh++) { ai[hh] += x * igw[kk * 4 + hh]; af[hh] += x * fgw[kk * 4 + hh]; }
  }
  __shared__ float sh[8 * 128];
  #pragma unroll
  for (int hh = 0; hh < 4; hh++) { sh[hh * 128 + tid] = ai[hh]; sh[(4 + hh) * 128 + tid] = af[hh]; }
  __syncthreads();
  for (int st = 64; st > 0; st >>= 1) {
    if (tid < st)
      #pragma unroll
      for (int a = 0; a < 8; a++) sh[a * 128 + tid] += sh[a * 128 + tid + st];
    __syncthreads();
  }
  if (tid < 4)          ig[((long long)(b * NHc + tid)) * Sc + s] = sh[tid * 128] + igb[tid];
  else if (tid < 8) { int hh = tid - 4; fg[((long long)(b * NHc + hh)) * Sc + s] = sh[tid * 128] + fgb[hh]; }
}

// ---- per-(b,head) scans: cs = cumsum(logsigmoid(fg)), rmax = runmax(ig-cs) ----
__global__ void mlstm_scan_kernel(const float* __restrict__ ig, const float* __restrict__ fg,
                                  float* __restrict__ cs, float* __restrict__ rm)
{
  int bn = blockIdx.x;
  int s  = threadIdx.x;             // 512
  __shared__ float sh[Sc];
  sh[s] = logsigf(fg[(long long)bn * Sc + s]);
  __syncthreads();
  for (int off = 1; off < Sc; off <<= 1) {
    float v = (s >= off) ? sh[s - off] : 0.f;
    __syncthreads();
    sh[s] += v;
    __syncthreads();
  }
  float css = sh[s];
  cs[(long long)bn * Sc + s] = css;
  __syncthreads();
  sh[s] = ig[(long long)bn * Sc + s] - css;
  __syncthreads();
  for (int off = 1; off < Sc; off <<= 1) {
    float v = (s >= off) ? sh[s - off] : -INFINITY;
    __syncthreads();
    sh[s] = fmaxf(sh[s], v);
    __syncthreads();
  }
  rm[(long long)bn * Sc + s] = sh[s];
}

// ---- stabilized gating + normalization of the QK^T rows (in place) ----
__global__ void attn_post_kernel(float* __restrict__ qk, const float* __restrict__ cs,
                                 const float* __restrict__ ig, const float* __restrict__ rm)
{
  int r  = blockIdx.x;              // (b*NH+n)*S + s
  int s  = r % Sc;
  int bn = r / Sc;
  const float inv = 0.0625f;        // 1/sqrt(256)
  float css = cs[(long long)bn * Sc + s];
  float rms = rm[(long long)bn * Sc + s];
  float* row = qk + (long long)r * Sc;
  __shared__ float red[256];
  int t0 = threadIdx.x, t1 = t0 + 256;
  float v0 = 0.f, v1 = 0.f;
  if (t0 <= s) v0 = row[t0] * inv * expf(ig[(long long)bn * Sc + t0] - cs[(long long)bn * Sc + t0] - rms);
  if (t1 <= s) v1 = row[t1] * inv * expf(ig[(long long)bn * Sc + t1] - cs[(long long)bn * Sc + t1] - rms);
  red[threadIdx.x] = v0 + v1;
  __syncthreads();
  for (int st = 128; st > 0; st >>= 1) {
    if (threadIdx.x < st) red[threadIdx.x] += red[threadIdx.x + st];
    __syncthreads();
  }
  float norm = fmaxf(fabsf(red[0]), expf(-(css + rms)));
  float dn = 1.f / (norm + EPSc);
  row[t0] = v0 * dn;
  row[t1] = v1 * dn;                 // zeros above diagonal
}

// ---- per-head layernorm of attention output -> (B,S,I) with norm weight ----
__global__ void headln_kernel(const float* __restrict__ ha, const float* __restrict__ nw,
                              float* __restrict__ hn)
{
  int r = blockIdx.x;               // (b*NH+n)*S + s
  int s = r % Sc;
  int bn = r / Sc;
  int n = bn % NHc, b = bn / NHc;
  const float* x = ha + (long long)r * DHMc;
  int d = threadIdx.x;              // 256
  float v = x[d];
  __shared__ float r1[256], r2[256];
  r1[d] = v; r2[d] = v * v;
  __syncthreads();
  for (int st = 128; st > 0; st >>= 1) {
    if (d < st) { r1[d] += r1[d + st]; r2[d] += r2[d + st]; }
    __syncthreads();
  }
  float mu = r1[0] / DHMc;
  float var = r2[0] / DHMc - mu * mu;
  float rs = rsqrtf(var + LNEPSc);
  hn[((long long)(b * Sc + s)) * Ic + n * DHMc + d] = (v - mu) * rs * nw[n * DHMc + d];
}

// ---- y = (hn + skip*xc) * swish(z)  (z = second half of up) ----
__global__ void mlstm_out_kernel(float* __restrict__ hn, const float* __restrict__ xc,
                                 const float* __restrict__ up, const float* __restrict__ skip)
{
  long long idx = (long long)blockIdx.x * blockDim.x + threadIdx.x;
  if (idx >= (long long)Bc * Sc * Ic) return;
  int c = (int)(idx & (Ic - 1));
  long long row = idx >> 10;
  float z = up[row * (2 * Ic) + Ic + c];
  hn[idx] = (hn[idx] + skip[c] * xc[idx]) * swishf(z);
}

// ------------------- sLSTM recurrence (per (b,head) block) -------------------
// 512 threads: thread t -> gate g=t>>7, out o=t&127. R (4,128,128) staged in
// 256KB dynamic LDS transposed as Rl[g][i][o] for broadcast/conflict-free MACs.
__global__ void __launch_bounds__(512) slstm_scan_kernel(
    const float* __restrict__ gx, const float* __restrict__ R,
    const float* __restrict__ bias, float* __restrict__ hs)
{
  extern __shared__ float Rl[];                  // 4*128*128 floats
  __shared__ float hb[DHSc];
  __shared__ float gsh[512];
  int bn = blockIdx.x;
  int n = bn & 3, b = bn >> 2;
  int tid = threadIdx.x;
  int g = tid >> 7, o = tid & 127;
  for (int e = tid; e < 4 * DHSc * DHSc; e += 512) {
    int gg = e >> 14;
    int oo = (e >> 7) & 127;
    int ii = e & 127;
    Rl[(gg * DHSc + ii) * DHSc + oo] = R[((long long)(gg * NHc + n) * DHSc + oo) * DHSc + ii];
  }
  float bv = bias[(g * NHc + n) * DHSc + o];
  if (tid < DHSc) hb[tid] = 0.f;
  float cst = 0.f, nst = 0.f, mst = 0.f;
  __syncthreads();
  const long long gstride = (long long)Bc * Sc * Ec;
  const float* rp = &Rl[g * DHSc * DHSc + o];
  for (int s = 0; s < Sc; s++) {
    float acc = gx[g * gstride + ((long long)(b * Sc + s)) * Ec + n * DHSc + o] + bv;
    #pragma unroll 4
    for (int i = 0; i < DHSc; i++) acc += hb[i] * rp[i * DHSc];
    gsh[tid] = acc;
    __syncthreads();
    if (tid < DHSc) {
      float ir  = gsh[o];
      float fr  = gsh[DHSc + o];
      float zr  = gsh[2 * DHSc + o];
      float orr = gsh[3 * DHSc + o];
      float lf = mst + logsigf(fr);
      float mn = fmaxf(ir, lf);
      float iv = expf(ir - mn), fv = expf(lf - mn);
      cst = fv * cst + iv * tanhf(zr);
      nst = fv * nst + iv;
      mst = mn;
      float hv = sigmf(orr) * cst / nst;
      hb[o] = hv;
      hs[((long long)(b * Sc + s)) * Ec + n * DHSc + o] = hv;
    }
    __syncthreads();
  }
}

// ---- sLSTM groupnorm over DHS per (b,s,head) + residual add into out ----
__global__ void slstm_gn_kernel(const float* __restrict__ hsv, const float* __restrict__ gw,
                                float* __restrict__ out)
{
  int r = blockIdx.x;               // (b*S+s)*NH + n
  int n = r & 3;
  long long bs = r >> 2;
  const float* x = hsv + bs * Ec + n * DHSc;
  int d = threadIdx.x;              // 128
  float v = x[d];
  __shared__ float r1[128], r2[128];
  r1[d] = v; r2[d] = v * v;
  __syncthreads();
  for (int st = 64; st > 0; st >>= 1) {
    if (d < st) { r1[d] += r1[d + st]; r2[d] += r2[d + st]; }
    __syncthreads();
  }
  float mu = r1[0] / DHSc;
  float var = r2[0] / DHSc - mu * mu;
  float rs = rsqrtf(var + LNEPSc);
  out[bs * Ec + n * DHSc + d] += (v - mu) * rs * gw[n * DHSc + d];
}

// ---- FFN: prod = gelu(a) * b with u = [a|b] along 2P ----
__global__ void gelumult_kernel(const float* __restrict__ u, float* __restrict__ p)
{
  long long idx = (long long)blockIdx.x * blockDim.x + threadIdx.x;
  if (idx >= (long long)Bc * Sc * Pc) return;
  int c = (int)(idx % Pc);
  long long row = idx / Pc;
  float a  = u[row * (2 * Pc) + c];
  float bg = u[row * (2 * Pc) + Pc + c];
  p[idx] = geluf(a) * bg;
}

// ===========================================================================
extern "C" void kernel_launch(void* const* d_in, const int* in_sizes, int n_in,
                              void* d_out, int out_size, void* d_ws, size_t ws_size,
                              hipStream_t stream)
{
  (void)in_sizes; (void)n_in; (void)out_size; (void)ws_size;
  const float* x_in    = (const float*)d_in[0];
  const float* fin_w   = (const float*)d_in[1];
  const float* fin_b   = (const float*)d_in[2];
  const float* m_ln_w  = (const float*)d_in[3];
  const float* m_up_w  = (const float*)d_in[4];
  const float* m_up_b  = (const float*)d_in[5];
  const float* m_conv_w= (const float*)d_in[6];
  const float* m_conv_b= (const float*)d_in[7];
  const float* m_q_w   = (const float*)d_in[8];
  const float* m_k_w   = (const float*)d_in[9];
  const float* m_v_w   = (const float*)d_in[10];
  const float* m_ig_w  = (const float*)d_in[11];
  const float* m_ig_b  = (const float*)d_in[12];
  const float* m_fg_w  = (const float*)d_in[13];
  const float* m_fg_b  = (const float*)d_in[14];
  const float* m_skip  = (const float*)d_in[15];
  const float* m_norm_w= (const float*)d_in[16];
  const float* m_down_w= (const float*)d_in[17];
  const float* m_down_b= (const float*)d_in[18];
  const float* s_ln_w  = (const float*)d_in[19];
  const float* s_conv_w= (const float*)d_in[20];
  const float* s_conv_b= (const float*)d_in[21];
  const float* s_gate_w= (const float*)d_in[22];
  const float* s_R     = (const float*)d_in[23];
  const float* s_bias  = (const float*)d_in[24];
  const float* s_gn_w  = (const float*)d_in[25];
  const float* s_ln2_w = (const float*)d_in[26];
  const float* s_ff_up_w  = (const float*)d_in[27];
  const float* s_ff_down_w= (const float*)d_in[28];
  const float* post_ln_w  = (const float*)d_in[29];

  float* out = (float*)d_out;                 // residual stream (B,S,E)
  float* ws  = (float*)d_ws;

  // workspace arena (floats); total ~44.1M floats (~176 MB)
  const size_t NBSE  = (size_t)Bc * Sc * Ec;        // 2,097,152
  const size_t NBS2I = (size_t)Bc * Sc * 2 * Ic;    // 8,388,608
  const size_t NBSI  = (size_t)Bc * Sc * Ic;        // 4,194,304
  const size_t NQK   = (size_t)Bc * NHc * Sc * Sc;  // 8,388,608
  const size_t NHA   = (size_t)Bc * NHc * Sc * DHMc;// 4,194,304
  const size_t NBNS  = (size_t)Bc * NHc * Sc;       // 16,384

  float* XIN = ws;
  float* UP  = XIN + NBSE;
  float* XC  = UP + NBS2I;
  float* Qb  = XC + NBSI;
  float* Kb  = Qb + NBSI;
  float* Vb  = Kb + NBSI;
  float* QK  = Vb + NBSI;
  float* HA  = QK + NQK;
  float* HN  = HA + NHA;
  float* IG  = HN + NBSI;
  float* FG  = IG + NBNS;
  float* CS  = FG + NBNS;
  float* RM  = CS + NBNS;
  // reuse:
  float* GX  = QK;   // 4*B*S*E == NQK
  float* SH  = Qb;   // B*S*E <= NBSI
  float* SXC = Kb;   // B*S*E <= NBSI
  float* FFU = UP;   // B*S*2P <= NBS2I
  float* FFP = XC;   // B*S*P  <= NBSI

  const int FLAG_WT = 1, FLAG_BIAS = 2, FLAG_ACC = 4;
  const int Mrows = Bc * Sc;   // 4096

  auto gemm = [&](const float* A, const float* Bw, const float* bias, float* C,
                  int M, int N, int Kd, int lda, int ldb, int ldc,
                  int batch, int dv,
                  long long sA1, long long sA2, long long sB1, long long sB2,
                  long long sC1, long long sC2, int flags) {
    dim3 g((N + 127) / 128, (M + 127) / 128, batch);
    gemm_bf16_wmma<<<g, dim3(256), 0, stream>>>(A, Bw, bias, C, M, N, Kd, lda, ldb, ldc,
                                                dv, sA1, sA2, sB1, sB2, sC1, sC2, flags);
  };

  // ffn_in: h = x @ fin_w + fin_b
  gemm(x_in, fin_w, fin_b, out, Mrows, Ec, Ec, Ec, Ec, Ec, 1, 1, 0,0,0,0,0,0, FLAG_BIAS);

  for (int l = 0; l < Lc; l++) {
    // -------- per-layer weight pointers --------
    const float* mlw  = m_ln_w  + (size_t)l * Ec;
    const float* muw  = m_up_w  + (size_t)l * Ec * 2 * Ic;
    const float* mub  = m_up_b  + (size_t)l * 2 * Ic;
    const float* mcw  = m_conv_w+ (size_t)l * Ic * KWc;
    const float* mcb  = m_conv_b+ (size_t)l * Ic;
    const float* mqw  = m_q_w   + (size_t)l * (Ic / 4) * 16;
    const float* mkw  = m_k_w   + (size_t)l * (Ic / 4) * 16;
    const float* mvw  = m_v_w   + (size_t)l * (Ic / 4) * 16;
    const float* migw = m_ig_w  + (size_t)l * 3 * Ic * NHc;
    const float* migb = m_ig_b  + (size_t)l * NHc;
    const float* mfgw = m_fg_w  + (size_t)l * 3 * Ic * NHc;
    const float* mfgb = m_fg_b  + (size_t)l * NHc;
    const float* mskp = m_skip  + (size_t)l * Ic;
    const float* mnw  = m_norm_w+ (size_t)l * Ic;
    const float* mdw  = m_down_w+ (size_t)l * Ic * Ec;
    const float* mdb  = m_down_b+ (size_t)l * Ec;
    const float* slw  = s_ln_w  + (size_t)l * Ec;
    const float* scw  = s_conv_w+ (size_t)l * Ec * KWc;
    const float* scb  = s_conv_b+ (size_t)l * Ec;
    const float* sgw  = s_gate_w+ (size_t)l * 4 * NHc * DHSc * DHSc;
    const float* sRl  = s_R     + (size_t)l * 4 * NHc * DHSc * DHSc;
    const float* sbl  = s_bias  + (size_t)l * 4 * NHc * DHSc;
    const float* sgnw = s_gn_w  + (size_t)l * Ec;
    const float* sl2w = s_ln2_w + (size_t)l * Ec;
    const float* sfu  = s_ff_up_w   + (size_t)l * Ec * 2 * Pc;
    const float* sfd  = s_ff_down_w + (size_t)l * Pc * Ec;

    // ================= mLSTM block =================
    ln_kernel<<<Mrows, 256, 0, stream>>>(out, mlw, XIN, Ec);
    gemm(XIN, muw, mub, UP, Mrows, 2 * Ic, Ec, Ec, 2 * Ic, 2 * Ic, 1, 1, 0,0,0,0,0,0, FLAG_BIAS);
    conv_swish_kernel<<<(unsigned)((NBSI + 255) / 256), 256, 0, stream>>>(UP, mcw, mcb, XC, Ic, 2 * Ic, (long long)NBSI);
    qkv_kernel<<<(unsigned)((NBSI + 255) / 256), 256, 0, stream>>>(XC, Ic, UP, 2 * Ic, mqw, mkw, mvw, Qb, Kb, Vb);
    igfg_kernel<<<Mrows, 128, 0, stream>>>(Qb, Kb, Vb, migw, migb, mfgw, mfgb, IG, FG);
    mlstm_scan_kernel<<<Bc * NHc, Sc, 0, stream>>>(IG, FG, CS, RM);
    // qk = qh @ kh^T  (batched over b,n; K stored (N,K) row-major -> wt)
    gemm(Qb, Kb, nullptr, QK, Sc, Sc, DHMc, Ic, Ic, Sc,
         Bc * NHc, NHc,
         (long long)Sc * Ic, DHMc, (long long)Sc * Ic, DHMc,
         (long long)NHc * Sc * Sc, (long long)Sc * Sc, FLAG_WT);
    attn_post_kernel<<<Bc * NHc * Sc, 256, 0, stream>>>(QK, CS, IG, RM);
    // h = Cn @ vh
    gemm(QK, Vb, nullptr, HA, Sc, DHMc, Sc, Sc, Ic, DHMc,
         Bc * NHc, NHc,
         (long long)NHc * Sc * Sc, (long long)Sc * Sc,
         (long long)Sc * Ic, DHMc,
         (long long)NHc * Sc * DHMc, (long long)Sc * DHMc, 0);
    headln_kernel<<<Bc * NHc * Sc, 256, 0, stream>>>(HA, mnw, HN);
    mlstm_out_kernel<<<(unsigned)((NBSI + 255) / 256), 256, 0, stream>>>(HN, XC, UP, mskp);
    gemm(HN, mdw, mdb, out, Mrows, Ec, Ic, Ic, Ec, Ec, 1, 1, 0,0,0,0,0,0, FLAG_BIAS | FLAG_ACC);

    // ================= sLSTM block =================
    ln_kernel<<<Mrows, 256, 0, stream>>>(out, slw, XIN, Ec);
    conv_swish_kernel<<<(unsigned)((NBSE + 255) / 256), 256, 0, stream>>>(XIN, scw, scb, SXC, Ec, Ec, (long long)NBSE);
    for (int g = 0; g < 4; g++) {
      const float* src = (g < 2) ? SXC : XIN;   // i,f from conv; z,o from pre-conv
      gemm(src, sgw + (size_t)g * NHc * DHSc * DHSc, nullptr, GX + (size_t)g * NBSE,
           Mrows, DHSc, DHSc, Ec, DHSc, Ec,
           NHc, NHc, 0, DHSc, 0, (long long)DHSc * DHSc, 0, DHSc, FLAG_WT);
    }
    slstm_scan_kernel<<<Bc * NHc, 512, 4 * DHSc * DHSc * sizeof(float), stream>>>(GX, sRl, sbl, SH);
    slstm_gn_kernel<<<Bc * Sc * NHc, 128, 0, stream>>>(SH, sgnw, out);

    // ================= gated FFN =================
    ln_kernel<<<Mrows, 256, 0, stream>>>(out, sl2w, XIN, Ec);
    gemm(XIN, sfu, nullptr, FFU, Mrows, 2 * Pc, Ec, Ec, 2 * Pc, 2 * Pc, 1, 1, 0,0,0,0,0,0, 0);
    gelumult_kernel<<<(unsigned)(((size_t)Bc * Sc * Pc + 255) / 256), 256, 0, stream>>>(FFU, FFP);
    gemm(FFP, sfd, nullptr, out, Mrows, Ec, Pc, Pc, Ec, Ec, 1, 1, 0,0,0,0,0,0, FLAG_ACC);
  }

  // final layernorm (in place)
  ln_kernel<<<Mrows, 256, 0, stream>>>(out, post_ln_w, out, Ec);
}
